// MultiHeadAttentionBlock_19988777796005
// MI455X (gfx1250) — matrix operand order repaired and checked by reading, not compiled, not run on hardware
//
#include <hip/hip_runtime.h>
#include <hip/hip_bf16.h>

// ---------------------------------------------------------------------------
// Problem dims (fixed by the reference)
// ---------------------------------------------------------------------------
#define BB 2
#define TT 1024
#define FF 1024
#define HH 16
#define DH 64
#define CC 4096
#define ND (HH*DH)        // 1024
#define SHIFT (CC-TT)     // 3072

typedef __attribute__((ext_vector_type(16))) __bf16          v16bf;
typedef __attribute__((ext_vector_type(8)))  float           v8f;
typedef __attribute__((ext_vector_type(8)))  unsigned short  us8;
typedef __attribute__((ext_vector_type(16))) unsigned short  us16;

#define LOG2E 1.44269504088896340736f

__device__ __forceinline__ unsigned short f2bf(float f) {
  unsigned int u = __builtin_bit_cast(unsigned int, f);
  u += 0x7fffu + ((u >> 16) & 1u);          // round-to-nearest-even
  return (unsigned short)(u >> 16);
}

// ---------------------------------------------------------------------------
// 1) f32 -> bf16 elementwise
// ---------------------------------------------------------------------------
__global__ void k_f32_to_bf16(const float* __restrict__ in,
                              unsigned short* __restrict__ out, int n) {
  int i = blockIdx.x * blockDim.x + threadIdx.x;
  if (i < n) out[i] = f2bf(in[i]);
}

// ---------------------------------------------------------------------------
// 2) transpose K x N f32 -> N x K bf16 (so WMMA B-fragments load contiguously)
// ---------------------------------------------------------------------------
__global__ void k_transpose_to_bf16(const float* __restrict__ in,   // K x N
                                    unsigned short* __restrict__ out, // N x K
                                    int K, int N) {
  int i = blockIdx.x * blockDim.x + threadIdx.x;
  if (i >= K * N) return;
  int n = i / K, k = i % K;
  out[i] = f2bf(in[(long)k * N + n]);
}

// ---------------------------------------------------------------------------
// 3) segment ids, next_position scan, new_mask (tiny, serial per batch)
// ---------------------------------------------------------------------------
__global__ void k_seg_prep(const int* __restrict__ mask,
                           const int* __restrict__ cache_mask,
                           const int* __restrict__ cache_position,
                           int* __restrict__ q_seg, int* __restrict__ k_seg,
                           int* __restrict__ out_next_pos,
                           int* __restrict__ out_new_mask) {
  int b = blockIdx.x * blockDim.x + threadIdx.x;
  if (b >= BB) return;
  int cs = 0;
  for (int i = 0; i < CC; ++i) {
    cs += cache_mask[b * CC + i];
    if (i >= TT) {                       // tail of cache cumsum -> k_seg[0..SHIFT)
      k_seg[b * CC + (i - TT)] = cs;
      out_new_mask[b * CC + (i - TT)] = cache_mask[b * CC + i];
    }
  }
  int Sb = cs;                            // max of cache cumsum == total
  int cm = 0;
  int carry = cache_position[b];
  for (int t = 0; t < TT; ++t) {
    int m = mask[b * TT + t];
    cm += m;
    q_seg[b * TT + t] = cm + Sb;
    k_seg[b * CC + SHIFT + t] = Sb + cm;
    out_new_mask[b * CC + SHIFT + t] = m;
    carry = m ? 0 : carry + 1;
  }
  out_next_pos[b] = carry;
}

// ---------------------------------------------------------------------------
// 4) bf16 GEMM: out[M,N] = A[M,K] * BT[N,K]^T + bias ; WMMA f32.16x16x32.bf16
//    4 waves per WG, each wave owns a 16x64 strip (4 accumulators).
//    Plain loop body on purpose: the compiler's pipeliner schedules this best
//    (loads one step ahead of the WMMAs, s_wait_loadcnt 0x1) — manual
//    prefetch/double-buffering was observed to defeat it.
// ---------------------------------------------------------------------------
__global__ void k_gemm_bf16(const unsigned short* __restrict__ A,
                            const unsigned short* __restrict__ BT,
                            const float* __restrict__ bias,
                            float* __restrict__ outF,
                            unsigned short* __restrict__ outB,
                            int M, int N, int K,
                            int rows_per_batch, long out_batch_stride,
                            long out_row_offset) {
  const int lane = threadIdx.x & 31;
  const int wave = threadIdx.x >> 5;                 // 0..3
  const int r0 = blockIdx.x * 64 + wave * 16;        // row base of this wave
  const int c0 = blockIdx.y * 64;                    // col base of this block
  const int nl = lane & 15;
  const int kbA = (lane < 16) ? 0 : 16;              // contiguous K layout (matches B)
  const int kbB = (lane < 16) ? 0 : 16;              // contiguous K layout (matches A)
  const int mA  = r0 + nl;

  v8f acc[4] = {v8f{}, v8f{}, v8f{}, v8f{}};

  const unsigned short* arow = A + (long)mA * K;
  for (int k0 = 0; k0 < K; k0 += 32) {
    us16 au;
    *(us8*)&au      = *(const us8*)(arow + k0 + kbA);       // K = kb..kb+7
    ((us8*)&au)[1]  = *(const us8*)(arow + k0 + kbA + 8);   // K = kb+8..kb+15
    v16bf af = __builtin_bit_cast(v16bf, au);
#pragma unroll
    for (int j = 0; j < 4; ++j) {
      const unsigned short* bp = BT + (long)(c0 + j * 16 + nl) * K + k0 + kbB;
      v16bf bf = __builtin_bit_cast(v16bf, *(const us16*)bp);
      acc[j] = __builtin_amdgcn_wmma_f32_16x16x32_bf16(
          false, af, false, bf, (short)0, acc[j], false, false);
    }
  }

  const int mbase = (lane < 16) ? 0 : 8;             // C/D layout row split
#pragma unroll
  for (int j = 0; j < 4; ++j) {
    int n = c0 + j * 16 + nl;
    float bv = bias ? bias[n] : 0.0f;
#pragma unroll
    for (int r = 0; r < 8; ++r) {
      int m = r0 + mbase + r;
      int bidx = m / rows_per_batch;
      int t    = m % rows_per_batch;
      long off = (long)bidx * out_batch_stride + out_row_offset +
                 (long)t * N + n;
      float val = acc[j][r] + bv;
      if (outF) outF[off] = val;
      if (outB) outB[off] = f2bf(val);
    }
  }
}

// ---------------------------------------------------------------------------
// 5) shift cache rows [TT..CC) -> [0..SHIFT) for K and V, f32 out + bf16 copy
// ---------------------------------------------------------------------------
__global__ void k_shift_cache(const float* __restrict__ ck,
                              const float* __restrict__ cv,
                              float* __restrict__ outK, float* __restrict__ outV,
                              unsigned short* __restrict__ kf,
                              unsigned short* __restrict__ vf) {
  long i = (long)blockIdx.x * blockDim.x + threadIdx.x;
  const long total = (long)BB * SHIFT * ND;
  if (i >= total) return;
  int n = (int)(i % ND);
  long rs = i / ND;
  int s = (int)(rs % SHIFT);
  int b = (int)(rs / SHIFT);
  long src = ((long)b * CC + (s + TT)) * ND + n;
  long dst = ((long)b * CC + s) * ND + n;
  float kv = ck[src], vv = cv[src];
  outK[dst] = kv;  outV[dst] = vv;
  kf[dst] = f2bf(kv);  vf[dst] = f2bf(vv);
}

// ---------------------------------------------------------------------------
// 6) V transpose: vt[b][h][d][s] = vf[b][s][h][d]  (d-major for P*V B-frags)
// ---------------------------------------------------------------------------
__global__ void k_transpose_v(const unsigned short* __restrict__ vf,
                              unsigned short* __restrict__ vt) {
  long i = (long)blockIdx.x * blockDim.x + threadIdx.x;
  const long total = (long)BB * HH * DH * CC;
  if (i >= total) return;
  int s = (int)(i % CC);
  long r = i / CC;
  int d = (int)(r % DH); r /= DH;
  int h = (int)(r % HH);
  int b = (int)(r / HH);
  vt[i] = vf[(((long)b * CC + s) * HH + h) * DH + d];
}

// ---------------------------------------------------------------------------
// 7) flash attention: 1 wave per (b, h, 16-query tile); 32-key chunks.
//    QK^T: 4 x v_wmma (2 N-tiles x 2 K-steps over Dh=64)
//    P*V : 4 x v_wmma (Dh split into 4 N-tiles), P relaid via LDS.
//    The softmax dependency chain dominates this loop, so branchless
//    speculative prefetch of the next chunk is pure upside here.
// ---------------------------------------------------------------------------
__global__ void k_attention(const unsigned short* __restrict__ qb,
                            const unsigned short* __restrict__ kf,
                            const unsigned short* __restrict__ vt,
                            const int* __restrict__ q_seg,
                            const int* __restrict__ k_seg,
                            unsigned short* __restrict__ attn) {
  __shared__ __align__(32) unsigned short ptile[16 * 32];
  const int lane = threadIdx.x;
  const int t0 = blockIdx.x * 16;
  const int h  = blockIdx.y;
  const int b  = blockIdx.z;
  const int nl = lane & 15;
  const int mbase = (lane < 16) ? 0 : 8;
  const int kbA = (lane < 16) ? 0 : 16;
  const int kbB = (lane < 16) ? 0 : 16;
  const float scale = 0.125f;  // 1/sqrt(Dh)

  // Q fragments (Dh = 64 -> two 16x32 A-fragments, constant over the loop)
  v16bf aq[2];
  {
    const unsigned short* qrow = qb + (((long)b * TT + (t0 + nl)) * HH + h) * DH;
#pragma unroll
    for (int kk = 0; kk < 2; ++kk) {
      us16 u;
      *(us8*)&u     = *(const us8*)(qrow + kk * 32 + kbA);
      ((us8*)&u)[1] = *(const us8*)(qrow + kk * 32 + kbA + 8);
      aq[kk] = __builtin_bit_cast(v16bf, u);
    }
  }
  int qs[8];
#pragma unroll
  for (int r = 0; r < 8; ++r) qs[r] = q_seg[(long)b * TT + t0 + mbase + r];

  v8f o[4] = {v8f{}, v8f{}, v8f{}, v8f{}};
  float mrow[8], lrow[8];
#pragma unroll
  for (int r = 0; r < 8; ++r) { mrow[r] = -1e30f; lrow[r] = 0.0f; }

  const int s_end = min(CC, t0 + 15 + SHIFT + 1);  // causal: s <= t + SHIFT
  for (int s0 = 0; s0 < s_end; s0 += 32) {
    // unconditional speculative prefetch of next chunk's K rows / V^T runs
    __builtin_prefetch(
        kf + (((long)b * CC + (s0 + 32 + nl)) * HH + h) * DH + kbB, 0, 3);
    __builtin_prefetch(
        vt + (((long)b * HH + h) * DH + nl) * CC + s0 + 32 + kbB, 0, 3);
    // ---- logits tile 16x32 ----
    v8f lacc[2] = {v8f{}, v8f{}};
#pragma unroll
    for (int jt = 0; jt < 2; ++jt) {
      int key = s0 + jt * 16 + nl;
      const unsigned short* krow = kf + (((long)b * CC + key) * HH + h) * DH;
#pragma unroll
      for (int kk = 0; kk < 2; ++kk) {
        v16bf bk = __builtin_bit_cast(v16bf, *(const us16*)(krow + kk * 32 + kbB));
        lacc[jt] = __builtin_amdgcn_wmma_f32_16x16x32_bf16(
            false, aq[kk], false, bk, (short)0, lacc[jt], false, false);
      }
    }
    // ---- mask + online softmax ----
    int ks0 = k_seg[(long)b * CC + s0 + nl];
    int ks1 = k_seg[(long)b * CC + s0 + 16 + nl];
    float p0[8], p1[8];
#pragma unroll
    for (int r = 0; r < 8; ++r) {
      int tq = t0 + mbase + r;
      int sA = s0 + nl, sB = s0 + 16 + nl;
      float la = ((qs[r] == ks0) && (sA <= tq + SHIFT)) ? lacc[0][r] * scale : -1.0e9f;
      float lb = ((qs[r] == ks1) && (sB <= tq + SHIFT)) ? lacc[1][r] * scale : -1.0e9f;
      float lm = fmaxf(la, lb);
#pragma unroll
      for (int x = 1; x < 16; x <<= 1) lm = fmaxf(lm, __shfl_xor(lm, x, 32));
      float mnew  = fmaxf(mrow[r], lm);
      float alpha = exp2f((mrow[r] - mnew) * LOG2E);
      float pa = exp2f((la - mnew) * LOG2E);
      float pb = exp2f((lb - mnew) * LOG2E);
      float rs = pa + pb;
#pragma unroll
      for (int x = 1; x < 16; x <<= 1) rs += __shfl_xor(rs, x, 32);
      lrow[r] = lrow[r] * alpha + rs;
      mrow[r] = mnew;
      p0[r] = pa; p1[r] = pb;
#pragma unroll
      for (int j = 0; j < 4; ++j) o[j][r] *= alpha;
    }
    // ---- relayout P (C/D layout -> A-fragment layout) via LDS ----
#pragma unroll
    for (int r = 0; r < 8; ++r) {
      ptile[(mbase + r) * 32 + nl]      = f2bf(p0[r]);
      ptile[(mbase + r) * 32 + 16 + nl] = f2bf(p1[r]);
    }
    __syncthreads();
    us16 pu;
    *(us8*)&pu     = *(const us8*)(&ptile[nl * 32 + kbA]);
    ((us8*)&pu)[1] = *(const us8*)(&ptile[nl * 32 + kbA + 8]);
    v16bf ap = __builtin_bit_cast(v16bf, pu);
    __syncthreads();
    // ---- O += P * V (V^T is d-major: contiguous key runs) ----
#pragma unroll
    for (int j = 0; j < 4; ++j) {
      int d = j * 16 + nl;
      const unsigned short* vp =
          vt + (((long)b * HH + h) * DH + d) * CC + s0 + kbB;
      v16bf bv = __builtin_bit_cast(v16bf, *(const us16*)vp);
      o[j] = __builtin_amdgcn_wmma_f32_16x16x32_bf16(
          false, ap, false, bv, (short)0, o[j], false, false);
    }
  }

  // ---- normalize + store attn (b,t,h,d) bf16 ----
#pragma unroll
  for (int r = 0; r < 8; ++r) {
    float inv = lrow[r] > 0.0f ? 1.0f / lrow[r] : 0.0f;
    int t = t0 + mbase + r;
#pragma unroll
    for (int j = 0; j < 4; ++j) {
      int d = j * 16 + nl;
      attn[(((long)b * TT + t) * HH + h) * DH + d] = f2bf(o[j][r] * inv);
    }
  }
}

// ---------------------------------------------------------------------------
// Host launcher
// ---------------------------------------------------------------------------
extern "C" void kernel_launch(void* const* d_in, const int* in_sizes, int n_in,
                              void* d_out, int out_size, void* d_ws,
                              size_t ws_size, hipStream_t stream) {
  (void)in_sizes; (void)n_in; (void)out_size; (void)ws_size;
  const float* x             = (const float*)d_in[0];
  const int*   mask          = (const int*)d_in[1];
  const float* cache_key     = (const float*)d_in[2];
  const float* cache_value   = (const float*)d_in[3];
  const int*   cache_mask    = (const int*)d_in[4];
  const int*   cache_position= (const int*)d_in[5];
  const float* Wq = (const float*)d_in[6];
  const float* bq = (const float*)d_in[7];
  const float* Wk = (const float*)d_in[8];
  const float* bk = (const float*)d_in[9];
  const float* Wv = (const float*)d_in[10];
  const float* bv = (const float*)d_in[11];
  const float* Wo = (const float*)d_in[12];
  const float* bo = (const float*)d_in[13];

  // d_out layout (4-byte slots): y | next_position | new_mask | k | v
  float* outY       = (float*)d_out;
  int*   outNextPos = (int*)((char*)d_out + (size_t)2097152 * 4);
  int*   outNewMask = (int*)((char*)d_out + (size_t)2097154 * 4);
  float* outK       = (float*)((char*)d_out + (size_t)2105346 * 4);
  float* outV       = (float*)((char*)d_out + (size_t)10493954 * 4);

  // workspace carve-up
  char* w = (char*)d_ws;
  auto carve = [&](size_t bytes) {
    char* p = w; w += (bytes + 255) & ~(size_t)255; return p;
  };
  unsigned short* xb   = (unsigned short*)carve((size_t)BB*TT*FF * 2);
  unsigned short* WqT  = (unsigned short*)carve((size_t)FF*ND * 2);
  unsigned short* WkT  = (unsigned short*)carve((size_t)FF*ND * 2);
  unsigned short* WvT  = (unsigned short*)carve((size_t)FF*ND * 2);
  unsigned short* WoT  = (unsigned short*)carve((size_t)ND*FF * 2);
  unsigned short* qbuf = (unsigned short*)carve((size_t)BB*TT*ND * 2);
  unsigned short* kful = (unsigned short*)carve((size_t)BB*CC*ND * 2);
  unsigned short* vful = (unsigned short*)carve((size_t)BB*CC*ND * 2);
  unsigned short* vtns = (unsigned short*)carve((size_t)BB*HH*DH*CC * 2);
  unsigned short* attn = (unsigned short*)carve((size_t)BB*TT*ND * 2);
  int* q_seg = (int*)carve((size_t)BB*TT * 4);
  int* k_seg = (int*)carve((size_t)BB*CC * 4);

  const int nx = BB * TT * FF;
  k_f32_to_bf16<<<(nx + 255) / 256, 256, 0, stream>>>(x, xb, nx);

  const int nw = FF * ND;
  k_transpose_to_bf16<<<(nw + 255) / 256, 256, 0, stream>>>(Wq, WqT, FF, ND);
  k_transpose_to_bf16<<<(nw + 255) / 256, 256, 0, stream>>>(Wk, WkT, FF, ND);
  k_transpose_to_bf16<<<(nw + 255) / 256, 256, 0, stream>>>(Wv, WvT, FF, ND);
  k_transpose_to_bf16<<<(nw + 255) / 256, 256, 0, stream>>>(Wo, WoT, ND, FF);

  k_seg_prep<<<1, 32, 0, stream>>>(mask, cache_mask, cache_position,
                                   q_seg, k_seg, outNextPos, outNewMask);

  // QKV projections: M=2048, N=1024, K=1024
  dim3 ggrid(BB * TT / 64, ND / 64);
  k_gemm_bf16<<<ggrid, 128, 0, stream>>>(xb, WqT, bq, (float*)nullptr, qbuf,
      BB * TT, ND, FF, TT, (long)TT * ND, 0L);
  k_gemm_bf16<<<ggrid, 128, 0, stream>>>(xb, WkT, bk, outK, kful,
      BB * TT, ND, FF, TT, (long)CC * ND, (long)SHIFT * ND);
  k_gemm_bf16<<<ggrid, 128, 0, stream>>>(xb, WvT, bv, outV, vful,
      BB * TT, ND, FF, TT, (long)CC * ND, (long)SHIFT * ND);

  const long nshift = (long)BB * SHIFT * ND;
  k_shift_cache<<<(int)((nshift + 255) / 256), 256, 0, stream>>>(
      cache_key, cache_value, outK, outV, kful, vful);

  const long nvt = (long)BB * HH * DH * CC;
  k_transpose_v<<<(int)((nvt + 255) / 256), 256, 0, stream>>>(vful, vtns);

  dim3 agrid(TT / 16, HH, BB);
  k_attention<<<agrid, 32, 0, stream>>>(qbuf, kful, vtns, q_seg, k_seg, attn);

  // Output projection: y = attn * Wo + bo
  dim3 ygrid(BB * TT / 64, FF / 64);
  k_gemm_bf16<<<ygrid, 128, 0, stream>>>(attn, WoT, bo, outY,
      (unsigned short*)nullptr, BB * TT, FF, ND, TT, (long)TT * FF, 0L);
}
